// BinaryMoSLinear_16054587752856
// MI455X (gfx1250) — matrix operand order repaired and checked by reading, not compiled, not run on hardware
//
#include <hip/hip_runtime.h>
#include <hip/hip_bf16.h>

typedef __attribute__((ext_vector_type(16))) __bf16 v16bf;
typedef __attribute__((ext_vector_type(8)))  float  v8f;

#define N_TOK 8192
#define H_DIM 4096
#define O_DIM 4096

#define BM 128
#define BN 128
#define BK 64
#define LDA (BK + 8)            // bf16 elements; 144-byte row stride keeps 16B alignment
#define A_BYTES (BM * LDA * 2)  // 18432
#define BUF_BYTES (2 * A_BYTES) // 36864 (A tile + B tile)
#define BUF_ELEMS (BUF_BYTES / 2)

// ---------------------------------------------------------------------------
// Kernel 1: router logits -> softmax -> fused input scaling -> bf16 activations
// ---------------------------------------------------------------------------
__global__ __launch_bounds__(256) void moe_route_scale(
    const float* __restrict__ x, const float* __restrict__ gate_w,
    const float* __restrict__ ics, float* __restrict__ rw,
    __hip_bfloat16* __restrict__ xs)
{
    const int n = blockIdx.x;
    const int tid = threadIdx.x;
    const float* xrow = x + (size_t)n * H_DIM;

    float p0 = 0.f, p1 = 0.f, p2 = 0.f, p3 = 0.f;
    for (int h = tid; h < H_DIM; h += 256) {
        float xv = xrow[h];
        p0 += xv * gate_w[h];
        p1 += xv * gate_w[H_DIM + h];
        p2 += xv * gate_w[2 * H_DIM + h];
        p3 += xv * gate_w[3 * H_DIM + h];
    }

    __shared__ float red[4][256];
    red[0][tid] = p0; red[1][tid] = p1; red[2][tid] = p2; red[3][tid] = p3;
    __syncthreads();
    for (int s = 128; s > 0; s >>= 1) {
        if (tid < s) {
            red[0][tid] += red[0][tid + s];
            red[1][tid] += red[1][tid + s];
            red[2][tid] += red[2][tid + s];
            red[3][tid] += red[3][tid + s];
        }
        __syncthreads();
    }

    __shared__ float rws[4];
    if (tid == 0) {
        float l0 = red[0][0], l1 = red[1][0], l2 = red[2][0], l3 = red[3][0];
        float m  = fmaxf(fmaxf(l0, l1), fmaxf(l2, l3));
        float e0 = __expf(l0 - m), e1 = __expf(l1 - m);
        float e2 = __expf(l2 - m), e3 = __expf(l3 - m);
        float inv = 1.0f / (e0 + e1 + e2 + e3);
        rws[0] = e0 * inv; rws[1] = e1 * inv; rws[2] = e2 * inv; rws[3] = e3 * inv;
        *(float4*)(rw + (size_t)n * 4) = make_float4(rws[0], rws[1], rws[2], rws[3]);
    }
    __syncthreads();

    const float w0 = rws[0], w1 = rws[1], w2 = rws[2], w3 = rws[3];
    __hip_bfloat16* xso = xs + (size_t)n * H_DIM;
    for (int h = tid; h < H_DIM; h += 256) {
        float is = w0 * ics[h] + w1 * ics[H_DIM + h] +
                   w2 * ics[2 * H_DIM + h] + w3 * ics[3 * H_DIM + h];
        xso[h] = __float2bfloat16(xrow[h] * is);
    }
}

// ---------------------------------------------------------------------------
// Kernel 2: binarize weights: sign(W) as exact bf16 bit patterns {-1,0,+1}
// ---------------------------------------------------------------------------
__global__ __launch_bounds__(256) void sign_to_bf16(
    const float* __restrict__ w, unsigned short* __restrict__ wb)
{
    const size_t i = ((size_t)blockIdx.x * 256 + threadIdx.x) * 8;
    float4 a = *(const float4*)(w + i);
    float4 b = *(const float4*)(w + i + 4);
    union { unsigned short s[8]; uint4 u; } o;
    o.s[0] = a.x > 0.f ? 0x3F80 : (a.x < 0.f ? 0xBF80 : 0);
    o.s[1] = a.y > 0.f ? 0x3F80 : (a.y < 0.f ? 0xBF80 : 0);
    o.s[2] = a.z > 0.f ? 0x3F80 : (a.z < 0.f ? 0xBF80 : 0);
    o.s[3] = a.w > 0.f ? 0x3F80 : (a.w < 0.f ? 0xBF80 : 0);
    o.s[4] = b.x > 0.f ? 0x3F80 : (b.x < 0.f ? 0xBF80 : 0);
    o.s[5] = b.y > 0.f ? 0x3F80 : (b.y < 0.f ? 0xBF80 : 0);
    o.s[6] = b.z > 0.f ? 0x3F80 : (b.z < 0.f ? 0xBF80 : 0);
    o.s[7] = b.w > 0.f ? 0x3F80 : (b.w < 0.f ? 0xBF80 : 0);
    *(uint4*)(wb + i) = o.u;
}

// ---------------------------------------------------------------------------
// Async 16-byte global->LDS copy (CDNA5 GLOBAL_LOAD_ASYNC_TO_LDS_B128).
// GVS mode: SGPR 64-bit base + per-lane 32-bit byte offset. VDST supplies the
// per-lane LDS byte offset (LDS_BASE is added by hardware). ASYNCcnt-tracked.
// ---------------------------------------------------------------------------
__device__ __forceinline__ void async_load16(unsigned ldsoff, unsigned gloff,
                                             const void* sbase)
{
    asm volatile("global_load_async_to_lds_b128 %0, %1, %2"
                 :: "v"(ldsoff), "v"(gloff), "s"(sbase)
                 : "memory");
}

// ---------------------------------------------------------------------------
// Kernel 3: y = (xs_bf16 @ wb_bf16^T) * (rw @ ocs) + bias   via WMMA bf16
// 128x128 block tile, BK=64, DOUBLE-BUFFERED LDS filled by async-to-LDS DMA.
// 8 waves: wave (r,c) in 4x2 owns a 32x64 subtile = 2x4 accumulator tiles.
// Last K-iteration peeled so the steady-state loop is branch-free.
// ---------------------------------------------------------------------------
__global__ __launch_bounds__(256) void bin_gemm_wmma(
    const unsigned short* __restrict__ xs,   // [N][H] bf16 bits
    const unsigned short* __restrict__ wb,   // [O][H] bf16 bits
    const float* __restrict__ rw,            // [N][4]
    const float* __restrict__ ocs,           // [4][O]
    const float* __restrict__ bias,          // [O]
    float* __restrict__ out)                 // [N][O] f32
{
    extern __shared__ unsigned short dynsmem[];   // 2 * (A tile + B tile)

    const int tid     = threadIdx.x;
    const int lane    = tid & 31;
    const int wid     = tid >> 5;
    const int waveRow = wid >> 1;      // 0..3 -> 32-row band
    const int waveCol = wid & 1;       // 0..1 -> 64-col band
    const int rowBase = blockIdx.y * BM;
    const int colBase = blockIdx.x * BN;

    const int lrow  = lane & 15;
    const int halfA = (lane >> 4) * 8;   // A K-offset: lanes 0-15 -> K{0-7,16-23}
    const int halfB = (lane >> 4) * 8;   // B K-offset: lanes 0-15 -> K{0-7,16-23}

    // Per-thread 16B-chunk offsets for tile staging (invariant across K: the
    // uniform kt*2 byte shift is folded into the async SGPR base address).
    unsigned ldsOff[4], aOff[4], bOff[4];
#pragma unroll
    for (int i = 0; i < 4; ++i) {
        int c  = tid + i * 256;          // 0..1023
        int r  = c >> 3;                 // 0..127
        int kc = (c & 7) * 8;            // 0,8,..,56  (bf16 elements)
        ldsOff[i] = (unsigned)(r * LDA + kc) * 2u;
        aOff[i]   = ((unsigned)(rowBase + r) * H_DIM + (unsigned)kc) * 2u;
        bOff[i]   = ((unsigned)(colBase + r) * H_DIM + (unsigned)kc) * 2u;
    }

    auto issue_tile = [&](int kt, int bsel) {
        const char* sA = (const char*)xs + (size_t)kt * 2;
        const char* sB = (const char*)wb + (size_t)kt * 2;
        const unsigned base = (unsigned)bsel * (unsigned)BUF_BYTES;
#pragma unroll
        for (int i = 0; i < 4; ++i)
            async_load16(base + ldsOff[i], aOff[i], sA);
#pragma unroll
        for (int i = 0; i < 4; ++i)
            async_load16(base + (unsigned)A_BYTES + ldsOff[i], bOff[i], sB);
    };

    // Prefetch tile 0 while we zero accumulators.
    issue_tile(0, 0);

    v8f acc[2][4];
#pragma unroll
    for (int rt = 0; rt < 2; ++rt)
#pragma unroll
        for (int ct = 0; ct < 4; ++ct)
#pragma unroll
            for (int e = 0; e < 8; ++e)
                acc[rt][ct][e] = 0.0f;

    union FragU { v16bf v; uint4 u[2]; };

    auto compute_tile = [&](int buf) {
        const unsigned short* lA = dynsmem + buf * BUF_ELEMS;
        const unsigned short* lB = lA + BM * LDA;
#pragma unroll
        for (int kk = 0; kk < BK; kk += 32) {
            FragU aF[2], bF[4];
#pragma unroll
            for (int rt = 0; rt < 2; ++rt) {
                int r = waveRow * 32 + rt * 16 + lrow;
                const unsigned short* p = &lA[r * LDA + kk + halfA];
                aF[rt].u[0] = *(const uint4*)p;          // K + {0..7}
                aF[rt].u[1] = *(const uint4*)(p + 16);   // K + {16..23}
            }
#pragma unroll
            for (int ct = 0; ct < 4; ++ct) {
                int cc = waveCol * 64 + ct * 16 + lrow;
                const unsigned short* p = &lB[cc * LDA + kk + halfB];
                bF[ct].u[0] = *(const uint4*)p;          // K + {0..7}
                bF[ct].u[1] = *(const uint4*)(p + 16);   // K + {16..23}
            }
#pragma unroll
            for (int rt = 0; rt < 2; ++rt)
#pragma unroll
                for (int ct = 0; ct < 4; ++ct)
                    acc[rt][ct] = __builtin_amdgcn_wmma_f32_16x16x32_bf16(
                        false, aF[rt].v, false, bF[ct].v,
                        (short)0, acc[rt][ct], false, false);
        }
    };

    // Steady state: branch-free; always issues the next tile into the other
    // buffer, then releases the current tile (loads complete in order, so
    // asynccnt <= 8 means the oldest 8 -- the current tile -- are done).
    int buf = 0;
    for (int kt = 0; kt < H_DIM - BK; kt += BK) {
        issue_tile(kt + BK, buf ^ 1);
        asm volatile("s_wait_asynccnt 0x8" ::: "memory");
        __syncthreads();   // current tile visible to all waves
        compute_tile(buf);
        __syncthreads();   // all reads of this buffer done before overwrite
        buf ^= 1;
    }

    // Peeled tail: nothing left to prefetch.
    asm volatile("s_wait_asynccnt 0x0" ::: "memory");
    __syncthreads();
    compute_tile(buf);

    // Epilogue: fuse rank-4 out_scale (rw[row] . ocs[:,col]) and bias.
    const int rhalf = (lane >> 4) << 3;   // C/D layout: lanes 16-31 hold M+8
#pragma unroll
    for (int ct = 0; ct < 4; ++ct) {
        const int gCol = colBase + waveCol * 64 + ct * 16 + lrow;
        const float o0 = ocs[gCol];
        const float o1 = ocs[O_DIM + gCol];
        const float o2 = ocs[2 * O_DIM + gCol];
        const float o3 = ocs[3 * O_DIM + gCol];
        const float bz = bias[gCol];
#pragma unroll
        for (int rt = 0; rt < 2; ++rt) {
#pragma unroll
            for (int r = 0; r < 8; ++r) {
                const int gRow = rowBase + waveRow * 32 + rt * 16 + r + rhalf;
                const float4 w4 = *(const float4*)(rw + (size_t)gRow * 4);
                const float os = w4.x * o0 + w4.y * o1 + w4.z * o2 + w4.w * o3;
                out[(size_t)gRow * O_DIM + gCol] = acc[rt][ct][r] * os + bz;
            }
        }
    }
}

// ---------------------------------------------------------------------------
extern "C" void kernel_launch(void* const* d_in, const int* in_sizes, int n_in,
                              void* d_out, int out_size, void* d_ws, size_t ws_size,
                              hipStream_t stream) {
    const float* x      = (const float*)d_in[0];   // [4,2048,4096]
    const float* weight = (const float*)d_in[1];   // [4096,4096]
    const float* bias   = (const float*)d_in[2];   // [4096]
    const float* gate_w = (const float*)d_in[3];   // [4,4096]
    const float* ics    = (const float*)d_in[4];   // [4,4096]
    const float* ocs    = (const float*)d_in[5];   // [4,4096]
    float* out = (float*)d_out;                    // [8192,4096] f32

    char* ws = (char*)d_ws;
    float*           rwbuf = (float*)ws;                                   // 128 KB
    __hip_bfloat16*  xsbuf = (__hip_bfloat16*)(ws + 131072);               // 64 MB
    unsigned short*  wbbuf = (unsigned short*)(ws + 131072 +
                                 (size_t)N_TOK * H_DIM * 2);               // 32 MB

    moe_route_scale<<<N_TOK, 256, 0, stream>>>(x, gate_w, ics, rwbuf, xsbuf);

    sign_to_bf16<<<(O_DIM * (size_t)H_DIM) / (256 * 8), 256, 0, stream>>>(weight, wbbuf);

    dim3 grid(O_DIM / BN, N_TOK / BM);   // 32 x 64 blocks
    bin_gemm_wmma<<<grid, 256, 2 * BUF_BYTES, stream>>>(
        (const unsigned short*)xsbuf, wbbuf, rwbuf, ocs, bias, out);
}
